// DGI_21414706938576
// MI455X (gfx1250) — compile-verified, matrix operand order checked by reading, not run on hardware
//
#include <hip/hip_runtime.h>

#define N_FEAT 512
#define N_HID  256

typedef __attribute__((ext_vector_type(16))) __bf16 v16bf;
typedef __attribute__((ext_vector_type(8)))  float  v8f;

__device__ __forceinline__ unsigned short f2bf(float f) {
  unsigned int u = __float_as_uint(f);
  u += 0x7FFFu + ((u >> 16) & 1u);          // round-to-nearest-even
  return (unsigned short)(u >> 16);
}

union BF16Frag {
  v16bf v;
  unsigned short s[16];
  uint4 q[2];
};

// Streaming f32 -> packed bf16 (8 elements per thread-iteration).
__global__ void f32_to_bf16_kernel(const float* __restrict__ in,
                                   unsigned short* __restrict__ outp, long n8) {
  long i = (long)blockIdx.x * blockDim.x + threadIdx.x;
  const long stride = (long)gridDim.x * blockDim.x;
  for (; i < n8; i += stride) {
    const float4* s = (const float4*)in + i * 2;
    const float4 a = s[0], b = s[1];
    union { uint4 q; unsigned short h[8]; } o;
    o.h[0] = f2bf(a.x); o.h[1] = f2bf(a.y); o.h[2] = f2bf(a.z); o.h[3] = f2bf(a.w);
    o.h[4] = f2bf(b.x); o.h[5] = f2bf(b.y); o.h[6] = f2bf(b.z); o.h[7] = f2bf(b.w);
    ((uint4*)outp)[i] = o.q;
  }
}

// h[m, n] = sum_k x[m,k] * w[n,k] + bias[n]   (x, w pre-converted to bf16)
// Block: 512 threads = 16 waves; block owns 16 M rows (A tile staged in LDS),
// each wave owns one 16-wide N tile. Inner loop: 2 ds_load_b128 + 2 global_load_b128 + 1 WMMA.
__global__ __launch_bounds__(512)
void gemm_bias_wmma(const unsigned short* __restrict__ xb,
                    const unsigned short* __restrict__ wb,
                    const float* __restrict__ bias, float* __restrict__ h,
                    int nNodes) {
  __shared__ unsigned short tileA[16 * N_FEAT];     // 16 KB
  const int tid  = threadIdx.x;
  const int lane = tid & 31;
  const int wave = tid >> 5;
  const int g    = lane >> 4;                       // lane group (0/1)
  const int r    = lane & 15;
  const int m0   = blockIdx.x * 16;

  // Cooperative A-tile load: 16 rows x 512 bf16 = 1024 uint4 (rows clamped at tail).
  {
    uint4* dst = (uint4*)tileA;
    for (int i = tid; i < 1024; i += 512) {
      const int row = i >> 6;                       // 64 uint4 per row
      const int off = i & 63;
      const int srcRow = min(m0 + row, nNodes - 1);
      dst[i] = ((const uint4*)(xb + (size_t)srcRow * N_FEAT))[off];
    }
  }
  __syncthreads();

  const int n0 = wave * 16;
  const unsigned short* __restrict__ brow = wb + (size_t)(n0 + r) * N_FEAT;
  const unsigned short* aL = tileA + r * N_FEAT;

  v8f c = {};
  #pragma unroll 4
  for (int kt = 0; kt < N_FEAT; kt += 32) {
    // lane's K runs inside this 32-block: [8g, 8g+7] and [16+8g, 23+8g]
    BF16Frag A, B;
    const uint4* ap = (const uint4*)(aL + kt + 8 * g);
    A.q[0] = ap[0];
    A.q[1] = ap[2];                                 // +16 bf16 = +2 uint4
    const uint4* bp = (const uint4*)(brow + kt + 8 * g);
    B.q[0] = bp[0];
    B.q[1] = bp[2];
    c = __builtin_amdgcn_wmma_f32_16x16x32_bf16(
        false, A.v, false, B.v, (short)0, c, false, false);
  }

  // C/D layout: lane -> n = n0 + r ; VGPR i -> m = m0 + i + 8*g
  const int n = n0 + r;
  const float bb = bias[n];
  float* __restrict__ hp = h + (size_t)(m0 + 8 * g) * N_HID + n;
  if (m0 + 16 <= nNodes) {                          // uniform fast path
    #pragma unroll
    for (int i = 0; i < 8; ++i) hp[(size_t)i * N_HID] = c[i] + bb;
  } else {
    #pragma unroll
    for (int i = 0; i < 8; ++i)
      if (m0 + 8 * g + i < nNodes) hp[(size_t)i * N_HID] = c[i] + bb;
  }
}

__global__ void zero_f4(float4* __restrict__ p, long n4) {
  long i = (long)blockIdx.x * blockDim.x + threadIdx.x;
  const long stride = (long)gridDim.x * blockDim.x;
  const float4 z = {0.f, 0.f, 0.f, 0.f};
  for (; i < n4; i += stride) p[i] = z;
}

// agg[row] += ev[e] * h[col];  one wave per edge, lane covers 8 features.
__global__ __launch_bounds__(256)
void spmm_kernel(const float* __restrict__ h, const int* __restrict__ ei,
                 const float* __restrict__ ev, float* __restrict__ agg,
                 int nEdges, int nNodes) {
  const int lane = threadIdx.x & 31;
  const int wave = threadIdx.x >> 5;
  const int e = blockIdx.x * 8 + wave;
  if (e >= nEdges) return;
  int row = ei[e];
  int col = ei[(size_t)nEdges + e];
  row = min(max(row, 0), nNodes - 1);
  col = min(max(col, 0), nNodes - 1);
  const float val = ev[e];
  const float4* __restrict__ src = (const float4*)(h + (size_t)col * N_HID) + lane * 2;
  const float4 p = src[0], q = src[1];
  float* dst = agg + (size_t)row * N_HID + lane * 8;
  unsafeAtomicAdd(dst + 0, val * p.x);
  unsafeAtomicAdd(dst + 1, val * p.y);
  unsafeAtomicAdd(dst + 2, val * p.z);
  unsafeAtomicAdd(dst + 3, val * p.w);
  unsafeAtomicAdd(dst + 4, val * q.x);
  unsafeAtomicAdd(dst + 5, val * q.y);
  unsafeAtomicAdd(dst + 6, val * q.z);
  unsafeAtomicAdd(dst + 7, val * q.w);
}

// In-place PReLU; optional per-column partial sums for the mean.
__global__ __launch_bounds__(256)
void prelu_colsum(float* __restrict__ agg, const float* __restrict__ prelu_a,
                  float* __restrict__ colsum, int nNodes, int doSum) {
  const int c = threadIdx.x;               // column 0..255
  const float a = prelu_a[0];
  const int n0 = blockIdx.x * 32;
  const int nEnd = min(n0 + 32, nNodes);
  float sum = 0.f;
  for (int n = n0; n < nEnd; ++n) {
    float* p = agg + (size_t)n * N_HID + c;
    float v = *p;
    v = (v >= 0.f) ? v : a * v;
    *p = v;
    sum += v;
  }
  if (doSum) unsafeAtomicAdd(&colsum[c], sum);
}

// s = sigmoid(colsum/N); v[h] = sum_g bil_w[h,g] * s[g]
__global__ __launch_bounds__(256)
void make_v_kernel(const float* __restrict__ colsum, const float* __restrict__ bilw,
                   float* __restrict__ v, float invN) {
  __shared__ float s[N_HID];
  const int t = threadIdx.x;
  const float m = colsum[t] * invN;
  s[t] = 1.f / (1.f + __expf(-m));
  __syncthreads();
  float acc = 0.f;
  const float* __restrict__ wr = bilw + (size_t)t * N_HID;
  #pragma unroll 8
  for (int g = 0; g < N_HID; ++g) acc += wr[g] * s[g];
  v[t] = acc;
}

// out[n] = dot(H[n,:], v) + bil_b ;  one wave per node, shuffle reduction.
__global__ __launch_bounds__(256)
void score_kernel(const float* __restrict__ H, const float* __restrict__ v,
                  const float* __restrict__ bil_b, float* __restrict__ out,
                  int nNodes) {
  const int lane = threadIdx.x & 31;
  const int wave = threadIdx.x >> 5;
  const int n = blockIdx.x * 8 + wave;
  if (n >= nNodes) return;
  const float4* __restrict__ hp = (const float4*)(H + (size_t)n * N_HID) + lane * 2;
  const float4* __restrict__ vp = (const float4*)v + lane * 2;
  const float4 h0 = hp[0], h1 = hp[1], v0 = vp[0], v1 = vp[1];
  float s = h0.x * v0.x + h0.y * v0.y + h0.z * v0.z + h0.w * v0.w
          + h1.x * v1.x + h1.y * v1.y + h1.z * v1.z + h1.w * v1.w;
  #pragma unroll
  for (int off = 16; off; off >>= 1) s += __shfl_xor(s, off, 32);
  if (lane == 0) out[n] = s + bil_b[0];
}

extern "C" void kernel_launch(void* const* d_in, const int* in_sizes, int n_in,
                              void* d_out, int out_size, void* d_ws, size_t ws_size,
                              hipStream_t stream) {
  const float* x1  = (const float*)d_in[0];
  const float* x2  = (const float*)d_in[1];
  const float* ev  = (const float*)d_in[2];
  const float* fcw = (const float*)d_in[3];
  const float* fcb = (const float*)d_in[4];
  const float* pa  = (const float*)d_in[5];
  const float* bw  = (const float*)d_in[6];
  const float* bb  = (const float*)d_in[7];
  const int*   ei  = (const int*)d_in[8];
  float* out = (float*)d_out;

  const int nNodes = in_sizes[0] / N_FEAT;   // 100000
  const int nEdges = in_sizes[2];            // 3200000
  const size_t hElems = (size_t)nNodes * N_HID;
  const size_t xElems = (size_t)nNodes * N_FEAT;
  const size_t wElems = (size_t)N_HID * N_FEAT;

  // workspace layout
  float* h_tmp  = (float*)d_ws;
  float* agg    = h_tmp + hElems;
  float* colsum = agg + hElems;              // 256 floats, contiguous after agg
  float* v      = colsum + N_HID;
  unsigned short* wb = (unsigned short*)(v + N_HID);
  unsigned short* xb = wb + wElems;

  const int gemmGrid  = (nNodes + 15) / 16;
  const int spmmGrid  = (nEdges + 7) / 8;
  const int preluGrid = (nNodes + 31) / 32;
  const int scoreGrid = (nNodes + 7) / 8;
  const long aggColsum4 = (long)(hElems + N_HID) / 4;
  const long agg4       = (long)hElems / 4;

  // convert weights once
  f32_to_bf16_kernel<<<64, 256, 0, stream>>>(fcw, wb, (long)(wElems / 8));

  // ---- pass 1: x_1 ----
  f32_to_bf16_kernel<<<8192, 256, 0, stream>>>(x1, xb, (long)(xElems / 8));
  gemm_bias_wmma<<<gemmGrid, 512, 0, stream>>>(xb, wb, fcb, h_tmp, nNodes);
  zero_f4<<<4096, 256, 0, stream>>>((float4*)agg, aggColsum4);   // agg + colsum
  spmm_kernel<<<spmmGrid, 256, 0, stream>>>(h_tmp, ei, ev, agg, nEdges, nNodes);
  prelu_colsum<<<preluGrid, 256, 0, stream>>>(agg, pa, colsum, nNodes, 1);
  make_v_kernel<<<1, 256, 0, stream>>>(colsum, bw, v, 1.0f / (float)nNodes);
  score_kernel<<<scoreGrid, 256, 0, stream>>>(agg, v, bb, out, nNodes);

  // ---- pass 2: x_2 (reuse buffers; v already computed from h_1) ----
  f32_to_bf16_kernel<<<8192, 256, 0, stream>>>(x2, xb, (long)(xElems / 8));
  gemm_bias_wmma<<<gemmGrid, 512, 0, stream>>>(xb, wb, fcb, h_tmp, nNodes);
  zero_f4<<<4096, 256, 0, stream>>>((float4*)agg, agg4);
  spmm_kernel<<<spmmGrid, 256, 0, stream>>>(h_tmp, ei, ev, agg, nEdges, nNodes);
  prelu_colsum<<<preluGrid, 256, 0, stream>>>(agg, pa, colsum, nNodes, 0);
  score_kernel<<<scoreGrid, 256, 0, stream>>>(agg, v, bb, out + nNodes, nNodes);
}
